// AudioEncoderV2_927712936521
// MI455X (gfx1250) — compile-verified, matrix operand order checked by reading, not run on hardware
//
#include <hip/hip_runtime.h>
#include <cstdint>

// ---------------- model constants ----------------
#define BATCH 2
#define NMELS 128
#define TIN   4096
#define T1    2048          // after conv1 stride 2
#define SEQ   1024          // after conv2 stride 2
#define DM    1280
#define NH    20
#define HD    64
#define DFF   5120
#define NL    4
#define KS    31
#define QSCALE 0.35355339059327373f   // 64^(-0.25)

typedef __attribute__((ext_vector_type(16))) __bf16 bf16x16;
typedef __attribute__((ext_vector_type(8)))  __bf16 bf16x8;
typedef __attribute__((ext_vector_type(8)))  float  f32x8;
typedef __attribute__((ext_vector_type(4)))  float  f32x4;

__device__ __forceinline__ f32x8 wmma_bf16(bf16x16 a, bf16x16 b, f32x8 c) {
  return __builtin_amdgcn_wmma_f32_16x16x32_bf16(false, a, false, b, (short)0, c,
                                                 false, false);
}
__device__ __forceinline__ float gelu_f(float x) {
  return 0.5f * x * (1.0f + erff(x * 0.70710678118654752f));
}
// Fragment (ISA 7.12.2): lane's 16 elems = two contiguous 16B runs at p and p+16.
__device__ __forceinline__ bf16x16 ld_frag_b(const __bf16* p) {
  bf16x8 lo = *(const bf16x8*)p;
  bf16x8 hi = *(const bf16x8*)(p + 16);
  bf16x16 r;
#pragma unroll
  for (int i = 0; i < 8; ++i) { r[i] = lo[i]; r[i + 8] = hi[i]; }
  return r;
}
__device__ __forceinline__ bf16x16 ld_frag_f(const float* p) {
  f32x8 lo = *(const f32x8*)p;
  f32x8 hi = *(const f32x8*)(p + 16);
  bf16x16 r;
#pragma unroll
  for (int i = 0; i < 8; ++i) { r[i] = (__bf16)lo[i]; r[i + 8] = (__bf16)hi[i]; }
  return r;
}

// =====================================================================
// Generic bf16 WMMA GEMM:  C[M,N](f32) = A[M,K](bf16,lda) * B[K,N](f32 wt)
// block tile 64x128, 8 waves of 32x32 (2x2 wmma tiles), K step 32.
// Double-buffered: A staged via GLOBAL_LOAD_ASYNC_TO_LDS_B128 overlapped
// with compute (ASYNCcnt pipelining); B prefetched to registers, stored
// transposed so fragments are contiguous b128 LDS loads.
// =====================================================================
#define TM 64
#define TN 128
#define TKK 32

template <bool CONV2B, bool BIAS, bool DOGELU, bool RES, bool STF, bool STB>
__global__ __launch_bounds__(256) void gemm_bf16_kernel(
    const __bf16* __restrict__ A, int lda, long aBatch,
    const float* __restrict__ Bw, const float* __restrict__ bias,
    const float* __restrict__ resid, float* __restrict__ Cf,
    __bf16* __restrict__ Cb, long cBatch, int M, int N, int K) {
  __shared__ __bf16 At[2][TM][TKK + 8];    // row-major, double buffered
  __shared__ __bf16 BtT[2][TN][TKK + 8];   // transposed [n][k], double buffered

  const int tid  = threadIdx.x;
  const int wave = tid >> 5, lane = tid & 31;
  const int l15 = lane & 15, kb = (lane >> 4) * 8;
  const int bz = blockIdx.z;
  const int m0 = blockIdx.y * TM;
  const int n0 = blockIdx.x * TN;
  const __bf16* Ab = A + (long)bz * aBatch;
  const int wm = (wave >> 2) * 32;   // 0 / 32
  const int wn = (wave & 3) * 32;    // 0..96

  // per-thread A async chunk: 64 rows x 4 x 16B per buffer
  const int ar = tid >> 2;
  const int ak = (tid & 3) * 8;
  const uint32_t a_lds[2] = {(uint32_t)(uintptr_t)&At[0][ar][ak],
                             (uint32_t)(uintptr_t)&At[1][ar][ak]};
  // per-thread B chunk (non-conv2): one row-run of 16 f32
  const int bkk = tid >> 3;
  const int bnc = (tid & 7) * 16;

  auto issueA = [&](int k0, int bufi) {
    const __bf16* gap = Ab + (long)(m0 + ar) * lda + (k0 + ak);
    asm volatile("global_load_async_to_lds_b128 %0, %1, off"
                 :: "v"(a_lds[bufi]), "v"((uint64_t)(uintptr_t)gap)
                 : "memory");
  };
  f32x4 breg[4];
  auto loadB = [&](int k0) {
    const float* bp = Bw + (long)(k0 + bkk) * N + (n0 + bnc);
    breg[0] = *(const f32x4*)(bp);
    breg[1] = *(const f32x4*)(bp + 4);
    breg[2] = *(const f32x4*)(bp + 8);
    breg[3] = *(const f32x4*)(bp + 12);
  };

  f32x8 acc[2][2];
#pragma unroll
  for (int i = 0; i < 2; ++i)
#pragma unroll
    for (int j = 0; j < 2; ++j) acc[i][j] = (f32x8){};

  // prologue: start tile 0 in flight
  issueA(0, 0);
  if (!CONV2B) loadB(0);

  int buf = 0;
  for (int k0 = 0; k0 < K; k0 += TKK, buf ^= 1) {
    // ---- commit B tile k0 (registers -> LDS, transposed) ----
    if (CONV2B) {
      for (int i = tid; i < TKK * TN; i += 256) {
        int kk = i >> 7, n = i & 127;
        int gk = k0 + kk, gn = n0 + n;
        int tap = gk / DM, ic = gk % DM;
        BtT[buf][n][kk] = (__bf16)Bw[((long)gn * DM + ic) * 3 + tap];
      }
    } else {
#pragma unroll
      for (int e = 0; e < 4; ++e) {
        BtT[buf][bnc + e][bkk]      = (__bf16)breg[0][e];
        BtT[buf][bnc + 4 + e][bkk]  = (__bf16)breg[1][e];
        BtT[buf][bnc + 8 + e][bkk]  = (__bf16)breg[2][e];
        BtT[buf][bnc + 12 + e][bkk] = (__bf16)breg[3][e];
      }
    }
    // ---- launch next tile's traffic, then wait only for tile k0 ----
    if (k0 + TKK < K) {
      issueA(k0 + TKK, buf ^ 1);
      if (!CONV2B) loadB(k0 + TKK);
      asm volatile("s_wait_asynccnt 0x1" ::: "memory");
    } else {
      asm volatile("s_wait_asynccnt 0x0" ::: "memory");
    }
    __syncthreads();

    bf16x16 bfr[2];
#pragma unroll
    for (int ct = 0; ct < 2; ++ct)
      bfr[ct] = ld_frag_b(&BtT[buf][wn + ct * 16 + l15][kb]);
#pragma unroll
    for (int rt = 0; rt < 2; ++rt) {
      bf16x16 afr = ld_frag_b(&At[buf][wm + rt * 16 + l15][kb]);
#pragma unroll
      for (int ct = 0; ct < 2; ++ct)
        acc[rt][ct] = wmma_bf16(afr, bfr[ct], acc[rt][ct]);
    }
    __syncthreads();
  }

  // epilogue
#pragma unroll
  for (int rt = 0; rt < 2; ++rt)
#pragma unroll
    for (int ct = 0; ct < 2; ++ct)
#pragma unroll
      for (int r = 0; r < 8; ++r) {
        int gm = m0 + wm + rt * 16 + r + (lane >> 4) * 8;
        int gn = n0 + wn + ct * 16 + l15;
        if (gm >= M || gn >= N) continue;
        float c = acc[rt][ct][r];
        if (BIAS) c += bias[gn];
        if (DOGELU) c = gelu_f(c);
        long idx = (long)bz * cBatch + (long)gm * N + gn;
        if (RES) c += resid[idx];
        if (STF) Cf[idx] = c;
        if (STB) Cb[idx] = (__bf16)c;
      }
}

// =====================================================================
// Attention: one workgroup = 64 queries for one (b,h).
// 64x1024 f32 score panel + Q tile + transposed V chunk in LDS (~273KB,
// sized for the CDNA5 320KB WGP LDS).
// =====================================================================
#define SF_STRIDE 1032
#define QS_OFF  (64 * SF_STRIDE * 4)
#define VT_OFF  (QS_OFF + 64 * 72 * 2)
#define RED_OFF (VT_OFF + 64 * 40 * 2)
#define ATT_SMEM_BYTES (RED_OFF + 256 * 4 + 64 * 4)

__global__ __launch_bounds__(256) void attention_kernel(
    const __bf16* __restrict__ qb, const __bf16* __restrict__ kbuf,
    const __bf16* __restrict__ vb, __bf16* __restrict__ attn,
    const int* __restrict__ x_len) {
  extern __shared__ char smem[];
  float(*Sf)[SF_STRIDE] = (float(*)[SF_STRIDE])smem;
  __bf16(*Qs)[72] = (__bf16(*)[72])(smem + QS_OFF);
  __bf16(*Vt)[40] = (__bf16(*)[40])(smem + VT_OFF);
  float* red  = (float*)(smem + RED_OFF);
  float* rinv = red + 256;

  const int tid  = threadIdx.x;
  const int lane = tid & 31, wave = tid >> 5;
  const int l15 = lane & 15, kb16 = (lane >> 4) * 8;
  const int q0 = blockIdx.x * 64;
  const int h  = blockIdx.y;
  const int b  = blockIdx.z;
  int len  = x_len[b];
  int len1 = (len - 1) / 2 + 1;
  int len2 = (len1 - 1) / 2 + 1;
  const long bh = (long)b * SEQ * DM + (long)h * HD;

  // stage Q tile (already RoPE'd + scaled), 16B vector copies
  for (int i = tid; i < 512; i += 256) {
    int r = i >> 3, d8 = (i & 7) * 8;
    *(bf16x8*)&Qs[r][d8] = *(const bf16x8*)(qb + bh + (long)(q0 + r) * DM + d8);
  }
  __syncthreads();

  // ---- pass 1: S = Q K^T. Q fragments hoisted: inner loop is
  //      4 global b128 loads + 8 wmma + score stores per key tile. ----
  {
    bf16x16 qfr[4][2];
#pragma unroll
    for (int rt = 0; rt < 4; ++rt) {
      int m = rt * 16 + l15;
      qfr[rt][0] = ld_frag_b(&Qs[m][kb16]);
      qfr[rt][1] = ld_frag_b(&Qs[m][32 + kb16]);
    }
    for (int ct = wave; ct < SEQ / 16; ct += 8) {
      int key0 = ct * 16;
      const __bf16* kp = kbuf + bh + (long)(key0 + l15) * DM + kb16;
      bf16x16 b0 = ld_frag_b(kp);
      bf16x16 b1 = ld_frag_b(kp + 32);
#pragma unroll
      for (int rt = 0; rt < 4; ++rt) {
        f32x8 acc = (f32x8){};
        acc = wmma_bf16(qfr[rt][0], b0, acc);
        acc = wmma_bf16(qfr[rt][1], b1, acc);
#pragma unroll
        for (int r = 0; r < 8; ++r) {
          int mm = rt * 16 + r + (lane >> 4) * 8;
          Sf[mm][key0 + l15] = acc[r];
        }
      }
    }
  }
  __syncthreads();

  // ---- pass 2: masked softmax (keys >= len2 excluded) ----
  {
    int row = tid >> 2, part = tid & 3;
    float mx = -3.0e38f;
    for (int c = part * 256; c < part * 256 + 256; ++c)
      if (c < len2) mx = fmaxf(mx, Sf[row][c]);
    red[row * 4 + part] = mx;
    __syncthreads();
    float rmx = fmaxf(fmaxf(red[row * 4 + 0], red[row * 4 + 1]),
                      fmaxf(red[row * 4 + 2], red[row * 4 + 3]));
    __syncthreads();
    float sm = 0.0f;
    for (int c = part * 256; c < part * 256 + 256; ++c) {
      float e = 0.0f;
      if (c < len2) e = expf(Sf[row][c] - rmx);
      Sf[row][c] = e;
      sm += e;
    }
    red[row * 4 + part] = sm;
    __syncthreads();
    if (part == 0) {
      float s = red[row * 4] + red[row * 4 + 1] + red[row * 4 + 2] + red[row * 4 + 3];
      rinv[row] = 1.0f / s;
    }
  }
  __syncthreads();

  // ---- pass 3: O = P V, workgroup-synchronous K loop, V staged transposed ----
  {
    const int t0 = wave, t1 = wave + 8;
    const int rt0 = t0 >> 2, ct0 = t0 & 3;
    const int rt1 = t1 >> 2, ct1 = t1 & 3;
    f32x8 acc0 = (f32x8){}, acc1 = (f32x8){};
    const int vkk = tid >> 3, vd8 = (tid & 7) * 8;
    for (int k0 = 0; k0 < SEQ; k0 += 32) {
      bf16x8 vrow = *(const bf16x8*)(vb + bh + (long)(k0 + vkk) * DM + vd8);
#pragma unroll
      for (int e = 0; e < 8; ++e) Vt[vd8 + e][vkk] = vrow[e];
      __syncthreads();
      bf16x16 a0 = ld_frag_f(&Sf[rt0 * 16 + l15][k0 + kb16]);
      bf16x16 b0 = ld_frag_b(&Vt[ct0 * 16 + l15][kb16]);
      acc0 = wmma_bf16(a0, b0, acc0);
      bf16x16 a1 = ld_frag_f(&Sf[rt1 * 16 + l15][k0 + kb16]);
      bf16x16 b1 = ld_frag_b(&Vt[ct1 * 16 + l15][kb16]);
      acc1 = wmma_bf16(a1, b1, acc1);
      __syncthreads();
    }
#pragma unroll
    for (int r = 0; r < 8; ++r) {
      int mm0 = rt0 * 16 + r + (lane >> 4) * 8;
      attn[bh + (long)(q0 + mm0) * DM + ct0 * 16 + l15] =
          (__bf16)(acc0[r] * rinv[mm0]);
      int mm1 = rt1 * 16 + r + (lane >> 4) * 8;
      attn[bh + (long)(q0 + mm1) * DM + ct1 * 16 + l15] =
          (__bf16)(acc1[r] * rinv[mm1]);
    }
  }
}

// =====================================================================
// conv1 (128ch -> 1280ch, k=3, s=2) + GELU, write bf16 (B,T1+2,DM),
// masked by len1 (== mask applied before conv2); pad rows pre-zeroed.
// =====================================================================
__global__ __launch_bounds__(256) void conv1_kernel(
    const float* __restrict__ x, const float* __restrict__ w,
    const float* __restrict__ bias, const int* __restrict__ x_len,
    __bf16* __restrict__ c1out) {
  long i = (long)blockIdx.x * 256 + threadIdx.x;
  if (i >= (long)BATCH * T1 * DM) return;
  int oc = (int)(i % DM);
  long r = i / DM;
  int t = (int)(r % T1);
  int b = (int)(r / T1);
  int len = x_len[b];
  int len1 = (len - 1) / 2 + 1;
  float acc = 0.0f;
  if (t < len1) {
    acc = bias[oc];
    const float* xb = x + (long)b * NMELS * TIN;
    const float* wr = w + (long)oc * NMELS * 3;
    for (int ic = 0; ic < NMELS; ++ic) {
      const float* xc = xb + (long)ic * TIN;
#pragma unroll
      for (int tap = 0; tap < 3; ++tap) {
        int tt = 2 * t + tap - 1;
        float xv = (tt >= 0 && tt < TIN && tt < len) ? xc[tt] : 0.0f;
        acc += xv * wr[ic * 3 + tap];
      }
    }
    acc = gelu_f(acc);
  }
  c1out[((long)b * (T1 + 2) + (t + 1)) * DM + oc] = (__bf16)acc;
}

// ---------------- LayerNorm (one block per row) ----------------
__global__ __launch_bounds__(256) void layernorm_kernel(
    const float* __restrict__ x, const float* __restrict__ g,
    const float* __restrict__ beta, __bf16* __restrict__ out, float eps) {
  __shared__ float red[256];
  long row = blockIdx.x;
  const float* xr = x + row * DM;
  float s = 0.0f;
  for (int i = threadIdx.x; i < DM; i += 256) s += xr[i];
  red[threadIdx.x] = s;
  __syncthreads();
  for (int off = 128; off > 0; off >>= 1) {
    if (threadIdx.x < off) red[threadIdx.x] += red[threadIdx.x + off];
    __syncthreads();
  }
  float mean = red[0] / DM;
  __syncthreads();
  float v = 0.0f;
  for (int i = threadIdx.x; i < DM; i += 256) {
    float d = xr[i] - mean;
    v += d * d;
  }
  red[threadIdx.x] = v;
  __syncthreads();
  for (int off = 128; off > 0; off >>= 1) {
    if (threadIdx.x < off) red[threadIdx.x] += red[threadIdx.x + off];
    __syncthreads();
  }
  float rstd = rsqrtf(red[0] / DM + eps);
  __syncthreads();
  for (int i = threadIdx.x; i < DM; i += 256)
    out[row * DM + i] = (__bf16)((xr[i] - mean) * rstd * g[i] + beta[i]);
}

// ---------------- RoPE + pre-scale (HD^-1/4), f32 -> bf16 ----------------
__global__ __launch_bounds__(256) void rope_kernel(const float* __restrict__ qf,
                                                   __bf16* __restrict__ qb) {
  long i = (long)blockIdx.x * 256 + threadIdx.x;
  if (i >= (long)BATCH * SEQ * NH * 32) return;
  int d = (int)(i % 32);
  long r = i / 32;
  int h = (int)(r % NH);
  long r2 = r / NH;
  int s = (int)(r2 % SEQ);
  int b = (int)(r2 / SEQ);
  float inv = powf(10000.0f, -(float)d / 32.0f);
  float f = (float)s * inv;
  float sn, cs;
  sincosf(f, &sn, &cs);
  long base = (((long)b * SEQ + s) * NH + h) * HD;
  float x1 = qf[base + d], x2 = qf[base + d + 32];
  qb[base + d]      = (__bf16)((x1 * cs - x2 * sn) * QSCALE);
  qb[base + d + 32] = (__bf16)((x2 * cs + x1 * sn) * QSCALE);
}

// ---------------- FSMN: 31-tap depthwise conv + residual, masked ----------------
__global__ __launch_bounds__(256) void fsmn_kernel(const float* __restrict__ vf,
                                                   const float* __restrict__ fw,
                                                   const int* __restrict__ x_len,
                                                   float* __restrict__ fsm) {
  long i = (long)blockIdx.x * 256 + threadIdx.x;
  if (i >= (long)BATCH * SEQ * DM) return;
  int d = (int)(i % DM);
  long r = i / DM;
  int t = (int)(r % SEQ);
  int b = (int)(r / SEQ);
  int len = x_len[b];
  int len1 = (len - 1) / 2 + 1;
  int len2 = (len1 - 1) / 2 + 1;
  float acc = 0.0f;
  if (t < len2) {
    const float* vbase = vf + (long)b * SEQ * DM + d;
    const float* wr = fw + (long)d * KS;
    for (int j = 0; j < KS; ++j) {
      int tt = t + j - 15;
      if (tt >= 0 && tt < SEQ && tt < len2) acc += vbase[(long)tt * DM] * wr[j];
    }
    acc += vbase[(long)t * DM];
  }
  fsm[i] = acc;
}

// ---------------- x += a + b; xb = bf16(x) ----------------
__global__ __launch_bounds__(256) void add3_kernel(float* __restrict__ x,
                                                   const float* __restrict__ a,
                                                   const float* __restrict__ bsrc,
                                                   __bf16* __restrict__ xb) {
  long i = (long)blockIdx.x * 256 + threadIdx.x;
  if (i >= (long)BATCH * SEQ * DM) return;
  float v = x[i] + a[i] + bsrc[i];
  x[i] = v;
  xb[i] = (__bf16)v;
}

// ---------------- write downsampled lengths (bit-exact int32) ----------------
__global__ void write_len_kernel(const int* __restrict__ x_len, int* __restrict__ o) {
  int b = threadIdx.x;
  if (b < BATCH) {
    int len = x_len[b];
    int l1 = (len - 1) / 2 + 1;
    o[b] = (l1 - 1) / 2 + 1;
  }
}

// =====================================================================
extern "C" void kernel_launch(void* const* d_in, const int* in_sizes, int n_in,
                              void* d_out, int out_size, void* d_ws, size_t ws_size,
                              hipStream_t stream) {
  const float* x_in    = (const float*)d_in[0];
  const int*   x_len   = (const int*)d_in[1];
  const float* conv1_w = (const float*)d_in[2];
  const float* conv1_b = (const float*)d_in[3];
  const float* conv2_w = (const float*)d_in[4];
  const float* conv2_b = (const float*)d_in[5];
  const float* q_w  = (const float*)d_in[6];
  const float* q_b  = (const float*)d_in[7];
  const float* k_w  = (const float*)d_in[8];
  const float* v_w  = (const float*)d_in[9];
  const float* v_b  = (const float*)d_in[10];
  const float* o_w  = (const float*)d_in[11];
  const float* o_b  = (const float*)d_in[12];
  const float* fsmn_w = (const float*)d_in[13];
  const float* ln1_s = (const float*)d_in[14];
  const float* ln1_b = (const float*)d_in[15];
  const float* w1 = (const float*)d_in[16];
  const float* b1 = (const float*)d_in[17];
  const float* w2 = (const float*)d_in[18];
  const float* b2 = (const float*)d_in[19];
  const float* ln2_s = (const float*)d_in[20];
  const float* ln2_b = (const float*)d_in[21];

  float* xout = (float*)d_out;                  // (B,SEQ,DM) result lives here
  int* len_out = (int*)(xout + (long)BATCH * SEQ * DM);

  char* wp = (char*)d_ws;
  auto alloc = [&](size_t bytes) {
    char* p = wp;
    wp += (bytes + 255) & ~(size_t)255;
    return p;
  };
  const long NSD = (long)BATCH * SEQ * DM;
  __bf16* c1out = (__bf16*)alloc((long)BATCH * (T1 + 2) * DM * 2);
  __bf16* xb    = (__bf16*)alloc(NSD * 2);
  __bf16* h     = (__bf16*)alloc(NSD * 2);
  float*  qf    = (float*)alloc(NSD * 4);
  float*  kf    = (float*)alloc(NSD * 4);
  float*  vf    = (float*)alloc(NSD * 4);
  __bf16* qbb   = (__bf16*)alloc(NSD * 2);
  __bf16* kbb   = (__bf16*)alloc(NSD * 2);
  __bf16* vbb   = (__bf16*)alloc(NSD * 2);
  __bf16* attn  = (__bf16*)alloc(NSD * 2);
  float*  tmp   = (float*)alloc(NSD * 4);
  float*  fsm   = (float*)alloc(NSD * 4);
  __bf16* mid   = (__bf16*)alloc((long)BATCH * SEQ * DFF * 2);

  const int EB = (int)((NSD + 255) / 256);
  dim3 blk(256);

  // ---- frontend ----
  hipMemsetAsync(c1out, 0, (long)BATCH * (T1 + 2) * DM * 2, stream);
  conv1_kernel<<<(int)(((long)BATCH * T1 * DM + 255) / 256), blk, 0, stream>>>(
      x_in, conv1_w, conv1_b, x_len, c1out);
  // conv2 as GEMM: per batch, A = c1out rows of 3840 contiguous bf16, lda = 2*DM
  gemm_bf16_kernel<true, true, true, false, true, true>
      <<<dim3(DM / TN, SEQ / TM, BATCH), blk, 0, stream>>>(
          c1out, 2 * DM, (long)(T1 + 2) * DM, conv2_w, conv2_b, nullptr, xout,
          xb, (long)SEQ * DM, SEQ, DM, 3 * DM);
  write_len_kernel<<<1, 32, 0, stream>>>(x_len, len_out);

  const int Mrows = BATCH * SEQ;   // 2048
  for (int l = 0; l < NL; ++l) {
    const float* qwl = q_w + (long)l * DM * DM;
    const float* kwl = k_w + (long)l * DM * DM;
    const float* vwl = v_w + (long)l * DM * DM;
    const float* owl = o_w + (long)l * DM * DM;
    const float* w1l = w1 + (long)l * DM * DFF;
    const float* w2l = w2 + (long)l * DFF * DM;

    layernorm_kernel<<<Mrows, blk, 0, stream>>>(xout, ln1_s + l * DM,
                                                ln1_b + l * DM, h, 1e-6f);
    // Q,K,V projections
    gemm_bf16_kernel<false, true, false, false, true, false>
        <<<dim3(DM / TN, Mrows / TM, 1), blk, 0, stream>>>(
            h, DM, 0, qwl, q_b + l * DM, nullptr, qf, nullptr, 0, Mrows, DM, DM);
    gemm_bf16_kernel<false, false, false, false, true, false>
        <<<dim3(DM / TN, Mrows / TM, 1), blk, 0, stream>>>(
            h, DM, 0, kwl, nullptr, nullptr, kf, nullptr, 0, Mrows, DM, DM);
    gemm_bf16_kernel<false, true, false, false, true, true>
        <<<dim3(DM / TN, Mrows / TM, 1), blk, 0, stream>>>(
            h, DM, 0, vwl, v_b + l * DM, nullptr, vf, vbb, 0, Mrows, DM, DM);
    // RoPE + scale
    {
      int rb = (int)(((long)BATCH * SEQ * NH * 32 + 255) / 256);
      rope_kernel<<<rb, blk, 0, stream>>>(qf, qbb);
      rope_kernel<<<rb, blk, 0, stream>>>(kf, kbb);
    }
    // attention (~273KB LDS: score panel + Q tile + V chunk)
    attention_kernel<<<dim3(SEQ / 64, NH, BATCH), blk, ATT_SMEM_BYTES, stream>>>(
        qbb, kbb, vbb, attn, x_len);
    // output projection
    gemm_bf16_kernel<false, true, false, false, true, false>
        <<<dim3(DM / TN, Mrows / TM, 1), blk, 0, stream>>>(
            attn, DM, 0, owl, o_b + l * DM, nullptr, tmp, nullptr, 0, Mrows, DM, DM);
    // FSMN branch
    fsmn_kernel<<<EB, blk, 0, stream>>>(vf, fsmn_w + (long)l * DM * KS, x_len, fsm);
    // residual: x += oproj + fsm
    add3_kernel<<<EB, blk, 0, stream>>>(xout, tmp, fsm, xb);
    // FFN
    layernorm_kernel<<<Mrows, blk, 0, stream>>>(xout, ln2_s + l * DM,
                                                ln2_b + l * DM, h, 1e-5f);
    gemm_bf16_kernel<false, true, true, false, false, true>
        <<<dim3(DFF / TN, Mrows / TM, 1), blk, 0, stream>>>(
            h, DM, 0, w1l, b1 + (long)l * DFF, nullptr, nullptr, mid, 0, Mrows,
            DFF, DM);
    gemm_bf16_kernel<false, true, false, true, true, true>
        <<<dim3(DM / TN, Mrows / TM, 1), blk, 0, stream>>>(
            mid, DFF, 0, w2l, b2 + l * DM, xout, xout, xb, 0, Mrows, DM, DFF);
  }
}
